// TopkSAGE_20744692039847
// MI455X (gfx1250) — matrix operand order repaired and checked by reading, not compiled, not run on hardware
//
#include <hip/hip_runtime.h>
#include <math.h>

typedef _Float16 f16;
typedef __attribute__((ext_vector_type(16))) _Float16 v16h;
typedef __attribute__((ext_vector_type(8)))  float    v8f;

#define HD   128
#define GC   32
#define ETOT 524288
#define NT0  32768
#define NEG_BIG (-3.402823466e38f)

// ---------------- utility fills ----------------
__global__ __launch_bounds__(256) void fill_f32(float* __restrict__ p, float v, int n) {
  int t = blockIdx.x * 256 + threadIdx.x;
  if (t < n) p[t] = v;
}
__global__ __launch_bounds__(256) void fill_i32(int* __restrict__ p, int v, int n) {
  int t = blockIdx.x * 256 + threadIdx.x;
  if (t < n) p[t] = v;
}

// ---------------- weight pre-pack into WMMA B-fragment order ----------------
// Bp[mat][kt][ct][lane][j] = W[kt*32 + (lane>>4)*16 + j][ct*16 + (lane&15)]  (f16)
__global__ __launch_bounds__(256)
void pack_weights_k(const float* __restrict__ Wl, const float* __restrict__ Wr,
                    f16* __restrict__ Bp) {
  int t = blockIdx.x * 256 + threadIdx.x;      // 32768 total
  int j    = t & 15;
  int lane = (t >> 4) & 31;
  int ct   = (t >> 9) & 7;
  int kt   = (t >> 12) & 3;
  int mat  = (t >> 14) & 1;
  int k = kt * 32 + (lane >> 4) * 16 + j;
  int n = ct * 16 + (lane & 15);
  const float* W = mat ? Wr : Wl;
  Bp[t] = (f16)W[k * HD + n];
}

// ---------------- edge aggregation (mean numerator + degree) ----------------
__global__ __launch_bounds__(256)
void aggregate_k(const float* __restrict__ h, const int* __restrict__ src,
                 const int* __restrict__ dst, const float* __restrict__ valid,
                 float* __restrict__ agg, float* __restrict__ cnt) {
  int t = blockIdx.x * 256 + threadIdx.x;
  int e = t >> 5;            // one wave per edge
  int lane = t & 31;         // lane owns 4 channels
  float v = valid[e];
  if (v == 0.0f) return;     // wave-uniform
  int s = src[e], d = dst[e];
  const float4 m = ((const float4*)(h + (size_t)s * HD))[lane];
  float* ap = agg + (size_t)d * HD + lane * 4;
  atomicAdd(ap + 0, m.x); atomicAdd(ap + 1, m.y);
  atomicAdd(ap + 2, m.z); atomicAdd(ap + 3, m.w);
  if (lane == 0) atomicAdd(&cnt[d], v);
}

__global__ __launch_bounds__(256)
void mean_div_k(float* __restrict__ agg, const float* __restrict__ cnt, int n) {
  int t = blockIdx.x * 256 + threadIdx.x;
  if (t < n) {
    int i = t >> 7;
    agg[t] = agg[t] / fmaxf(cnt[i], 1.0f);
  }
}

// ---------------- fused SAGE linear: h = mean@Wl + hin@Wr + bb (WMMA) ----------------
__global__ __launch_bounds__(256)
void sage_gemm_k(const float* __restrict__ mean, const float* __restrict__ hin,
                 const f16* __restrict__ Bp, const float* __restrict__ bb,
                 float* __restrict__ hout) {
  __shared__ f16 sA[2][16][136];               // 136-half pitch: conflict-free
  int row0 = blockIdx.x * 16;
  int tid = threadIdx.x;
  for (int i = tid; i < 16 * HD; i += 256) {
    int r = i >> 7, c = i & 127;
    sA[0][r][c] = (f16)mean[(size_t)(row0 + r) * HD + c];
    sA[1][r][c] = (f16)hin [(size_t)(row0 + r) * HD + c];
  }
  __syncthreads();
  int wave = tid >> 5, lane = tid & 31;
  int m = lane & 15, khalf = lane >> 4;
  v8f acc = {};
#pragma unroll
  for (int mat = 0; mat < 2; ++mat) {
#pragma unroll
    for (int kt = 0; kt < 4; ++kt) {
      int kb = kt * 32;
      v16h a, b;
      b = *(const v16h*)(Bp + ((((mat * 4 + kt) * 8 + wave) * 32 + lane) << 4));
#pragma unroll
      for (int j = 0; j < 16; ++j) {
        int kk = khalf * 16 + j;   // contiguous k order, matches packed B
        a[j] = sA[mat][m][kb + kk];
      }
      acc = __builtin_amdgcn_wmma_f32_16x16x32_f16(false, a, false, b,
                                                   (short)0, acc, false, false);
    }
  }
  int n = wave * 16 + m;
  float bias = bb[n];
#pragma unroll
  for (int r = 0; r < 8; ++r)
    hout[(size_t)(row0 + r + khalf * 8) * HD + n] = acc[r] + bias;
}

// ---------------- batch-norm ----------------
__global__ __launch_bounds__(128)
void bn_stats_k(const float* __restrict__ h, float* __restrict__ sums,
                float* __restrict__ sqs, int nt) {
  int c = threadIdx.x;
  int r0 = blockIdx.x * 32;
  int r1 = r0 + 32; if (r1 > nt) r1 = nt;
  float s = 0.0f, q = 0.0f;
  for (int r = r0; r < r1; ++r) {
    float v = h[(size_t)r * HD + c];
    s += v; q += v * v;
  }
  atomicAdd(&sums[c], s);
  atomicAdd(&sqs[c], q);
}

__global__ __launch_bounds__(128)
void bn_finalize_k(const float* __restrict__ sums, const float* __restrict__ sqs,
                   const float* __restrict__ g, const float* __restrict__ be,
                   float ntf, float* __restrict__ ca, float* __restrict__ cb) {
  int c = threadIdx.x;
  float mu  = sums[c] / ntf;
  float var = sqs[c] / ntf - mu * mu;            // biased variance
  float rstd = rsqrtf(var + 1e-5f);
  float a = rstd * g[c];
  ca[c] = a;
  cb[c] = be[c] - mu * a;
}

__global__ __launch_bounds__(256)
void bn_apply_relu_k(float* __restrict__ h, const float* __restrict__ ca,
                     const float* __restrict__ cb, int n) {
  int t = blockIdx.x * 256 + threadIdx.x;
  if (t < n) {
    int c = t & 127;
    h[t] = fmaxf(h[t] * ca[c] + cb[c], 0.0f);
  }
}

// ---------------- top-k scoring ----------------
__global__ void pnorm_k(const float* __restrict__ p, float* __restrict__ pn) {
  int lane = threadIdx.x;                        // 32
  float4 pv = ((const float4*)p)[lane];
  float s = pv.x * pv.x + pv.y * pv.y + pv.z * pv.z + pv.w * pv.w;
#pragma unroll
  for (int o = 16; o > 0; o >>= 1) s += __shfl_xor(s, o, 32);
  if (lane == 0) pn[0] = sqrtf(s);
}

__global__ __launch_bounds__(256)
void score_k(const float* __restrict__ h, const float* __restrict__ p,
             const float* __restrict__ pn, float* __restrict__ score, int nt) {
  int t = blockIdx.x * 256 + threadIdx.x;
  int node = t >> 5, lane = t & 31;
  if (node >= nt) return;
  float4 hv = ((const float4*)(h + (size_t)node * HD))[lane];
  float4 pv = ((const float4*)p)[lane];
  float s = hv.x * pv.x + hv.y * pv.y + hv.z * pv.z + hv.w * pv.w;
#pragma unroll
  for (int o = 16; o > 0; o >>= 1) s += __shfl_xor(s, o, 32);
  if (lane == 0) score[node] = s / pn[0];
}

// per-graph bitonic top-k (descending, index tie-break)
__global__ __launch_bounds__(1024)
void topk_k(const float* __restrict__ score, int ncur, int k,
            int* __restrict__ perm, float* __restrict__ gate) {
  __shared__ float key[1024];
  __shared__ int   sid[1024];
  int g = blockIdx.x, t = threadIdx.x;
  if (t < ncur) { key[t] = score[g * ncur + t]; sid[t] = t; }
  else          { key[t] = NEG_BIG;            sid[t] = (1 << 20) + t; }
  __syncthreads();
  for (int size = 2; size <= 1024; size <<= 1) {
    for (int stride = size >> 1; stride > 0; stride >>= 1) {
      int p = t ^ stride;
      if (p > t) {
        float ka = key[t], kb = key[p];
        int   ia = sid[t], ib = sid[p];
        bool aFirst = (ka > kb) || (ka == kb && ia < ib);
        bool desc = ((t & size) == 0);
        if (desc ? !aFirst : aFirst) {
          key[t] = kb; key[p] = ka; sid[t] = ib; sid[p] = ia;
        }
      }
      __syncthreads();
    }
  }
  if (t < k) {
    perm[g * k + t] = g * ncur + sid[t];
    gate[g * k + t] = tanhf(key[t]);
  }
}

__global__ __launch_bounds__(256)
void scatter_map_k(const int* __restrict__ perm, int* __restrict__ mapping, int nk) {
  int t = blockIdx.x * 256 + threadIdx.x;
  if (t < nk) mapping[perm[t]] = t;
}

__global__ __launch_bounds__(256)
void gather_gate_k(const float* __restrict__ h, const int* __restrict__ perm,
                   const float* __restrict__ gate, float* __restrict__ out, int n) {
  int t = blockIdx.x * 256 + threadIdx.x;
  if (t < n) {
    int i = t >> 7, c = t & 127;
    out[t] = h[(size_t)perm[i] * HD + c] * gate[i];
  }
}

__global__ __launch_bounds__(256)
void remap_k(const int* __restrict__ src, const int* __restrict__ dst,
             const float* __restrict__ valid, const int* __restrict__ mapping,
             int* __restrict__ nsrc, int* __restrict__ ndst, float* __restrict__ nvalid) {
  int e = blockIdx.x * 256 + threadIdx.x;
  int s = mapping[src[e]], d = mapping[dst[e]];
  float v = valid[e];
  if (s < 0 || d < 0) v = 0.0f;
  nsrc[e] = s < 0 ? 0 : s;
  ndst[e] = d < 0 ? 0 : d;
  nvalid[e] = v;
}

// ---------------- readout + MLP head ----------------
__global__ __launch_bounds__(128)
void readout_k(const float* __restrict__ hp, int k, float* __restrict__ flat, int off) {
  int g = blockIdx.x, c = threadIdx.x;
  float s = 0.0f, mx = NEG_BIG;
  for (int j = 0; j < k; ++j) {
    float v = hp[((size_t)g * k + j) * HD + c];
    s += v; if (v > mx) mx = v;
  }
  flat[g * 1024 + off + c] = s;
  flat[g * 1024 + off + 128 + c] = mx;
}

__global__ __launch_bounds__(256)
void dense1_k(const float* __restrict__ flat, const float* __restrict__ W,
              const float* __restrict__ b, float* __restrict__ hd) {
  int t = blockIdx.x * 256 + threadIdx.x;     // 32*512
  int g = t >> 9, o = t & 511;
  float s = b[o];
  for (int i = 0; i < 1024; ++i) s = fmaf(flat[g * 1024 + i], W[i * 512 + o], s);
  hd[t] = fmaxf(s, 0.0f);
}

__global__ __launch_bounds__(64)
void dense2_k(const float* __restrict__ hd, const float* __restrict__ W,
              const float* __restrict__ b, float* __restrict__ out) {
  int t = blockIdx.x * 64 + threadIdx.x;      // 320
  if (t >= GC * 10) return;
  int g = t / 10, o = t % 10;
  float s = b[o];
  for (int i = 0; i < 512; ++i) s = fmaf(hd[g * 512 + i], W[i * 10 + o], s);
  out[t] = s;
}

// ---------------- driver ----------------
extern "C" void kernel_launch(void* const* d_in, const int* in_sizes, int n_in,
                              void* d_out, int out_size, void* d_ws, size_t ws_size,
                              hipStream_t stream) {
  (void)in_sizes; (void)n_in; (void)out_size; (void)ws_size;
  const float* x  = (const float*)d_in[0];
  const int*   ei = (const int*)d_in[1];
  const float *Wl[4], *Wr[4], *bbv[4], *gv[4], *bev[4], *pv[4];
  for (int l = 0; l < 4; ++l) {
    int base = 3 + l * 6;
    Wl[l]  = (const float*)d_in[base + 0];
    Wr[l]  = (const float*)d_in[base + 1];
    bbv[l] = (const float*)d_in[base + 2];
    gv[l]  = (const float*)d_in[base + 3];
    bev[l] = (const float*)d_in[base + 4];
    pv[l]  = (const float*)d_in[base + 5];
  }
  const float* Wd1 = (const float*)d_in[27];
  const float* bd1 = (const float*)d_in[28];
  const float* Wd2 = (const float*)d_in[29];
  const float* bd2 = (const float*)d_in[30];
  float* out = (float*)d_out;

  char* base = (char*)d_ws;
  size_t off = 0;
  auto alloc = [&](size_t bytes) -> void* {
    void* p = base + off;
    off += (bytes + 255) & ~(size_t)255;
    return p;
  };
  float* hA      = (float*)alloc((size_t)NT0 * HD * 4);
  float* hB      = (float*)alloc((size_t)NT0 * HD * 4);
  float* agg     = (float*)alloc((size_t)NT0 * HD * 4);
  float* cnt     = (float*)alloc((size_t)NT0 * 4);
  float* score   = (float*)alloc((size_t)NT0 * 4);
  int*   mapping = (int*)  alloc((size_t)NT0 * 4);
  int*   perm    = (int*)  alloc((size_t)NT0 * 4);
  float* gate    = (float*)alloc((size_t)NT0 * 4);
  int*   srcA    = (int*)  alloc((size_t)ETOT * 4);
  int*   dstA    = (int*)  alloc((size_t)ETOT * 4);
  float* valA    = (float*)alloc((size_t)ETOT * 4);
  int*   srcB    = (int*)  alloc((size_t)ETOT * 4);
  int*   dstB    = (int*)  alloc((size_t)ETOT * 4);
  float* valB    = (float*)alloc((size_t)ETOT * 4);
  f16*   Bp      = (f16*)  alloc(32768 * 2);
  float* bnacc   = (float*)alloc(256 * 4);     // [0:128) sums, [128:256) sumsq
  float* bncoef  = (float*)alloc(256 * 4);     // [0:128) scale, [128:256) shift
  float* pn      = (float*)alloc(256);
  float* flat    = (float*)alloc((size_t)GC * 1024 * 4);
  float* hd      = (float*)alloc((size_t)GC * 512 * 4);

  hipMemcpyAsync(hA, x, (size_t)NT0 * HD * 4, hipMemcpyDeviceToDevice, stream);
  hipMemcpyAsync(srcA, ei, (size_t)ETOT * 4, hipMemcpyDeviceToDevice, stream);
  hipMemcpyAsync(dstA, ei + ETOT, (size_t)ETOT * 4, hipMemcpyDeviceToDevice, stream);
  fill_f32<<<ETOT / 256, 256, 0, stream>>>(valA, 1.0f, ETOT);

  int ncur = 1024;
  for (int l = 0; l < 4; ++l) {
    int nt = GC * ncur;                  // 32768, 26240, 20992, 16800 (all %16==0)
    int k  = (4 * ncur + 4) / 5;         // ceil(0.8*ncur): 820, 656, 525, 420
    int nk = GC * k;

    pack_weights_k<<<128, 256, 0, stream>>>(Wl[l], Wr[l], Bp);
    fill_f32<<<(nt * HD) / 256, 256, 0, stream>>>(agg, 0.0f, nt * HD);
    fill_f32<<<(nt + 255) / 256, 256, 0, stream>>>(cnt, 0.0f, nt);
    aggregate_k<<<(ETOT * 32) / 256, 256, 0, stream>>>(hA, srcA, dstA, valA, agg, cnt);
    mean_div_k<<<(nt * HD) / 256, 256, 0, stream>>>(agg, cnt, nt * HD);
    sage_gemm_k<<<nt / 16, 256, 0, stream>>>(agg, hA, Bp, bbv[l], hB);

    fill_f32<<<1, 256, 0, stream>>>(bnacc, 0.0f, 256);
    bn_stats_k<<<(nt + 31) / 32, 128, 0, stream>>>(hB, bnacc, bnacc + 128, nt);
    bn_finalize_k<<<1, 128, 0, stream>>>(bnacc, bnacc + 128, gv[l], bev[l],
                                         (float)nt, bncoef, bncoef + 128);
    bn_apply_relu_k<<<(nt * HD) / 256, 256, 0, stream>>>(hB, bncoef, bncoef + 128, nt * HD);

    pnorm_k<<<1, 32, 0, stream>>>(pv[l], pn);
    score_k<<<(nt * 32) / 256, 256, 0, stream>>>(hB, pv[l], pn, score, nt);
    topk_k<<<GC, 1024, 0, stream>>>(score, ncur, k, perm, gate);
    gather_gate_k<<<(nk * HD + 255) / 256, 256, 0, stream>>>(hB, perm, gate, hA, nk * HD);
    readout_k<<<GC, 128, 0, stream>>>(hA, k, flat, l * 256);

    if (l < 3) {
      fill_i32<<<(nt + 255) / 256, 256, 0, stream>>>(mapping, -1, nt);
      scatter_map_k<<<(nk + 255) / 256, 256, 0, stream>>>(perm, mapping, nk);
      remap_k<<<ETOT / 256, 256, 0, stream>>>(srcA, dstA, valA, mapping, srcB, dstB, valB);
      int* ti;
      ti = srcA; srcA = srcB; srcB = ti;
      ti = dstA; dstA = dstB; dstB = ti;
      float* tf = valA; valA = valB; valB = tf;
    }
    ncur = k;
  }

  dense1_k<<<(GC * 512) / 256, 256, 0, stream>>>(flat, Wd1, bd1, hd);
  dense2_k<<<(GC * 10 + 63) / 64, 64, 0, stream>>>(hd, Wd2, bd2, out);
}
